// SwinTransformerBlock_kv_20366734917794
// MI455X (gfx1250) — matrix operand order repaired and checked by reading, not compiled, not run on hardware
//
#include <hip/hip_runtime.h>
#include <hip/hip_bf16.h>

// ---------------- CDNA5 WMMA types & fragment loaders ----------------
typedef float  v8f   __attribute__((ext_vector_type(8)));
typedef __bf16 v16bf __attribute__((ext_vector_type(16)));
typedef __bf16 v8bf  __attribute__((ext_vector_type(8)));

__device__ __forceinline__ v16bf cat16(v8bf lo, v8bf hi) {
  return __builtin_shufflevector(lo, hi, 0,1,2,3,4,5,6,7,8,9,10,11,12,13,14,15);
}
// A-matrix 16x32 bf16 fragment: lane = (khalf<<4)|m.
// VGPR0-7 hold K = khalf*16 .. +15 (contiguous per half)
__device__ __forceinline__ v16bf load_fragA(const __bf16* base, int half) {
  v8bf lo = *(const v8bf*)(base + half * 16);
  v8bf hi = *(const v8bf*)(base + half * 16 + 8);
  return cat16(lo, hi);
}
// B-matrix 32x16 bf16 fragment: lane = (khalf<<4)|n, K contiguous per half.
__device__ __forceinline__ v16bf load_fragB(const __bf16* base, int half) {
  v8bf lo = *(const v8bf*)(base + half * 16);
  v8bf hi = *(const v8bf*)(base + half * 16 + 8);
  return cat16(lo, hi);
}
__device__ __forceinline__ v8f wmma_bf16(v16bf a, v16bf b, v8f c) {
  return __builtin_amdgcn_wmma_f32_16x16x32_bf16(false, a, false, b, (short)0, c, false, false);
}

#define NTOK  343
#define NPAD  352
#define CCH   192
#define NHEAD 6
#define HDIM  32
#define NWIN  128
#define LTOK  43904      /* = NWIN*NTOK = S*H*W */
#define HIDN  768
#define QSCALE 0.17677669529663687f  /* 32^-0.5 */
#define NEGBIG (-3.0e38f)

// ---------------- weight transpose + bf16 convert ----------------
__global__ void prep_weights(const float* __restrict__ kv_w, const float* __restrict__ proj_w,
                             const float* __restrict__ fc1_w, const float* __restrict__ fc2_w,
                             __bf16* __restrict__ kvT, __bf16* __restrict__ projT,
                             __bf16* __restrict__ fc1T, __bf16* __restrict__ fc2T) {
  int idx = blockIdx.x * 256 + threadIdx.x;
  if (idx < 73728) {                       // kv: 384*192
    int o = idx / 192, i = idx % 192;
    kvT[idx] = (__bf16)kv_w[i * 384 + o];
  } else if (idx < 73728 + 36864) {        // proj: 192*192
    int t = idx - 73728; int o = t / 192, i = t % 192;
    projT[t] = (__bf16)proj_w[i * 192 + o];
  } else if (idx < 73728 + 36864 + 147456) { // fc1: 768*192
    int t = idx - 73728 - 36864; int o = t / 192, i = t % 192;
    fc1T[t] = (__bf16)fc1_w[i * 768 + o];
  } else if (idx < 73728 + 36864 + 147456 + 147456) { // fc2: 192*768
    int t = idx - 73728 - 36864 - 147456; int o = t / 768, i = t % 768;
    fc2T[t] = (__bf16)fc2_w[i * 192 + o];
  }
}

// ---------------- pre-gather relative-position bias -> (head, n, mpad) f32, 0-padded ----
__global__ void prep_bias(const float* __restrict__ relb, float* __restrict__ biasT) {
  int idx = blockIdx.x * 256 + threadIdx.x;
  if (idx >= NHEAD * NTOK * NPAD) return;
  int head = idx / (NTOK * NPAD);
  int t = idx - head * (NTOK * NPAD);
  int qn = t / NPAD, mcol = t - qn * NPAD;
  float v = 0.f;
  if (mcol < NTOK) {
    int qsl = qn / 49, qhl = (qn / 7) % 7, qwl = qn % 7;
    int msl = mcol / 49, mhl = (mcol / 7) % 7, mwl = mcol % 7;
    int ridx = (qhl - mhl + 6) * 20 + (qsl - msl + 6) * 13 + (qwl - mwl + 6);
    v = relb[ridx * NHEAD + head];
  }
  biasT[idx] = v;
}

// ---------------- pre-pad mask -> (win, n, mpad) f32, -3e38 in the key padding --------
__global__ void prep_mask(const float* __restrict__ mask, float* __restrict__ maskP) {
  long idx = (long)blockIdx.x * 256 + threadIdx.x;
  if (idx >= (long)NWIN * NTOK * NPAD) return;
  int win = (int)(idx / (NTOK * NPAD));
  int t = (int)(idx - (long)win * (NTOK * NPAD));
  int qn = t / NPAD, mcol = t - qn * NPAD;
  float v = NEGBIG;
  if (mcol < NTOK) v = mask[((long)win * NTOK + qn) * NTOK + mcol];
  maskP[idx] = v;
}

// ---------------- LN + cyclic shift + window partition ----------------
__global__ void ln_shift_part(const float* __restrict__ skip, const float* __restrict__ xup,
                              const float* __restrict__ g, const float* __restrict__ b,
                              __bf16* __restrict__ sk_w, __bf16* __restrict__ xu_w) {
  int wave = threadIdx.x >> 5, lane = threadIdx.x & 31;
  long tok = (long)blockIdx.x * 4 + wave;          // [0, NWIN*NTOK)
  int win = (int)(tok / NTOK), n = (int)(tok % NTOK);
  int ws_s = win >> 6, ws_h = (win >> 3) & 7, ws_w = win & 7;
  int ls = n / 49, lh = (n / 7) % 7, lw = n % 7;
  int so = (ws_s * 7 + ls + 3) % 14;
  int ho = (ws_h * 7 + lh + 3) % 56;
  int wo = (ws_w * 7 + lw + 3) % 56;
  long l = ((long)so * 56 + ho) * 56 + wo;
  const float* ps = skip + l * CCH;
  const float* pu = xup + l * CCH;
  float vs[6], vu[6], ss = 0.f, su = 0.f;
  for (int i = 0; i < 6; ++i) {
    vs[i] = ps[i * 32 + lane]; vu[i] = pu[i * 32 + lane];
    ss += vs[i]; su += vu[i];
  }
  for (int o = 16; o; o >>= 1) { ss += __shfl_xor(ss, o); su += __shfl_xor(su, o); }
  float ms = ss / 192.f, mu = su / 192.f, qs = 0.f, qu = 0.f;
  for (int i = 0; i < 6; ++i) {
    float d = vs[i] - ms; qs += d * d;
    d = vu[i] - mu; qu += d * d;
  }
  for (int o = 16; o; o >>= 1) { qs += __shfl_xor(qs, o); qu += __shfl_xor(qu, o); }
  float rs = rsqrtf(qs / 192.f + 1e-5f), ru = rsqrtf(qu / 192.f + 1e-5f);
  __bf16* os = sk_w + tok * CCH;
  __bf16* ou = xu_w + tok * CCH;
  for (int i = 0; i < 6; ++i) {
    int c = i * 32 + lane;
    float gc = g[c], bc = b[c];
    os[c] = (__bf16)((vs[i] - ms) * rs * gc + bc);
    ou[c] = (__bf16)(((vu[i] - mu) * ru * gc + bc) * QSCALE);
  }
}

// ---------------- plain LN (norm2) -> bf16 ----------------
__global__ void ln_plain(const float* __restrict__ x, const float* __restrict__ g,
                         const float* __restrict__ b, __bf16* __restrict__ out) {
  int wave = threadIdx.x >> 5, lane = threadIdx.x & 31;
  long tok = (long)blockIdx.x * 4 + wave;
  const float* p = x + tok * CCH;
  float v[6], s = 0.f;
  for (int i = 0; i < 6; ++i) { v[i] = p[i * 32 + lane]; s += v[i]; }
  for (int o = 16; o; o >>= 1) s += __shfl_xor(s, o);
  float m = s / 192.f, q = 0.f;
  for (int i = 0; i < 6; ++i) { float d = v[i] - m; q += d * d; }
  for (int o = 16; o; o >>= 1) q += __shfl_xor(q, o);
  float r = rsqrtf(q / 192.f + 1e-5f);
  __bf16* po = out + tok * CCH;
  for (int i = 0; i < 6; ++i) {
    int c = i * 32 + lane;
    po[c] = (__bf16)((v[i] - m) * r * g[c] + b[c]);
  }
}

// ---------------- generic WMMA GEMM: C[M,N] = A[M,K] * Bt[N,K] + bias ----------------
// MODE 0: store bf16            MODE 1: gelu -> bf16
// MODE 2: + add -> f32 out      MODE 3: proj scatter (reverse+roll) + shortcut -> f32
template <int MODE>
__global__ void gemm_bf16(const __bf16* __restrict__ A, const __bf16* __restrict__ Bt,
                          const float* __restrict__ bias, int M, int Nc, int K,
                          __bf16* __restrict__ outb, float* __restrict__ outf,
                          const float* __restrict__ add) {
  int wave = threadIdx.x >> 5, lane = threadIdx.x & 31;
  int tile = blockIdx.x * (blockDim.x >> 5) + wave;
  int mt = M >> 4, nt = Nc >> 4;
  if (tile >= mt * nt) return;
  int tM = tile % mt, tN = tile / mt;
  int half = lane >> 4, ln = lane & 15;
  const __bf16* a0 = A + (long)(tM * 16 + ln) * K;
  const __bf16* b0 = Bt + (long)(tN * 16 + ln) * K;
  v8f acc = {0.f, 0.f, 0.f, 0.f, 0.f, 0.f, 0.f, 0.f};
  for (int kb = 0; kb < K; kb += 32) {
    v16bf af = load_fragA(a0 + kb, half);
    v16bf bfr = load_fragB(b0 + kb, half);
    acc = wmma_bf16(af, bfr, acc);
  }
  int col = tN * 16 + ln;
  float bc = bias[col];
  for (int r = 0; r < 8; ++r) {
    int row = tM * 16 + r + 8 * half;
    float v = acc[r] + bc;
    if (MODE == 0) {
      outb[(long)row * Nc + col] = (__bf16)v;
    } else if (MODE == 1) {
      float gv = 0.5f * v * (1.f + erff(v * 0.70710678118654752f));
      outb[(long)row * Nc + col] = (__bf16)gv;
    } else if (MODE == 2) {
      long o = (long)row * Nc + col;
      outf[o] = v + add[o];
    } else { // MODE 3: row = win*343 + n -> flat token via reverse + roll(+3)
      int win = row / NTOK, n = row - win * NTOK;
      int ws_s = win >> 6, ws_h = (win >> 3) & 7, ws_w = win & 7;
      int ls = n / 49, lh = (n / 7) % 7, lw = n % 7;
      int so = (ws_s * 7 + ls + 3) % 14;
      int ho = (ws_h * 7 + lh + 3) % 56;
      int wo = (ws_w * 7 + lw + 3) % 56;
      long o = (((long)so * 56 + ho) * 56 + wo) * CCH + col;
      outf[o] = v + add[o];
    }
  }
}

// ---------------- windowed attention: one block per (window, head) ----------------
// LDS: K[352][32] bf16 | Vt[32][352] bf16 | per-wave scores 16x352 f32 |
//      per-wave P 16x352 bf16 | per-wave denom[16] f32  (~180 KB of the 320 KB WGP LDS)
__global__ void attn_kernel(const __bf16* __restrict__ xu_w, const __bf16* __restrict__ kvbuf,
                            const float* __restrict__ biasT, const float* __restrict__ maskP,
                            __bf16* __restrict__ attnout) {
  extern __shared__ char smem[];
  __bf16* Ksm   = (__bf16*)smem;                  // 22528 B
  __bf16* Vtsm  = (__bf16*)(smem + 22528);        // 22528 B
  float*  scAll = (float*)(smem + 45056);         // 4*16*352*4 = 90112 B
  __bf16* pAll  = (__bf16*)(smem + 135168);       // 4*16*352*2 = 45056 B
  float*  dAll  = (float*)(smem + 180224);        // 4*16*4 = 256 B

  int blk = blockIdx.x, win = blk / NHEAD, head = blk % NHEAD;
  int tid = threadIdx.x, wave = tid >> 5, lane = tid & 31;
  int half = lane >> 4, ln = lane & 15;
  const __bf16* kvw = kvbuf + (long)win * NTOK * 384;
  const v8bf z8 = {(__bf16)0.f,(__bf16)0.f,(__bf16)0.f,(__bf16)0.f,
                   (__bf16)0.f,(__bf16)0.f,(__bf16)0.f,(__bf16)0.f};

  // stage K rows (row-major, d contiguous) with zero padding to 352
  for (int idx = tid; idx < NPAD * 4; idx += 128) {
    int m = idx >> 2, c = idx & 3;
    v8bf val = z8;
    if (m < NTOK) val = *(const v8bf*)(kvw + (long)m * 384 + head * HDIM + c * 8);
    *(v8bf*)(Ksm + m * HDIM + c * 8) = val;
  }
  // stage V transposed: Vt[d][m], m contiguous
  for (int d = 0; d < HDIM; ++d) {
    for (int m = tid; m < NPAD; m += 128) {
      __bf16 bv = (__bf16)0.f;
      if (m < NTOK) bv = kvw[(long)m * 384 + CCH + head * HDIM + d];
      Vtsm[d * NPAD + m] = bv;
    }
  }
  __syncthreads();

  float*  scW = scAll + wave * 16 * NPAD;
  __bf16* pW  = pAll + wave * 16 * NPAD;
  float*  dW  = dAll + wave * 16;
  const __bf16* qbase = xu_w + (long)win * NTOK * CCH + head * HDIM;
  const float* biasH = biasT + (long)head * NTOK * NPAD;   // [qn][mpad], 0-padded
  const float* maskW = maskP + (long)win * NTOK * NPAD;    // [qn][mpad], -3e38-padded

  // lane-invariant single-base LDS pointers -> constant-offset ds_load_b128s
  const __bf16* kfrag0 = Ksm + ln * HDIM + half * 16;          // + mt2*512 elems
  const __bf16* pfrag0 = pW + ln * NPAD + half * 16;           // + kb elems (hi: +8)
  float* scr0 = scW + (8 * half) * NPAD;                       // + r*NPAD + mcol

  for (int qt = wave; qt < 22; qt += 4) {
    int q0 = qt * 16;
    int qrow = q0 + ln; if (qrow > NTOK - 1) qrow = NTOK - 1;
    v16bf qf = load_fragA(qbase + (long)qrow * CCH, half);   // Q 16x32 (pre-scaled)

    // raw scores = q k^T   (22 key tiles; one base reg + immediate DS offsets)
    for (int mt2 = 0; mt2 < 22; ++mt2) {
      v8bf klo = *(const v8bf*)(kfrag0 + mt2 * 16 * HDIM);
      v8bf khi = *(const v8bf*)(kfrag0 + mt2 * 16 * HDIM + 8);
      v8f acc = {0.f, 0.f, 0.f, 0.f, 0.f, 0.f, 0.f, 0.f};
      acc = wmma_bf16(qf, cat16(klo, khi), acc);
      int mcol = mt2 * 16 + ln;
      for (int r = 0; r < 8; ++r) scr0[r * NPAD + mcol] = acc[r];
    }
    // bias + mask + softmax over contiguous row halves (lane ln owns row i)
    {
      int i = ln;
      int qn = q0 + i; if (qn > NTOK - 1) qn = NTOK - 1;   // padded rows: discarded later
      float* srow = scW + i * NPAD + half * 176;
      const float* brow = biasH + (long)qn * NPAD + half * 176;
      const float* mrow = maskW + (long)qn * NPAD + half * 176;
      float mx = NEGBIG;
      for (int c = 0; c < 176; ++c) {
        float sc = srow[c] + brow[c] + mrow[c];   // padding cols -> ~-3e38
        srow[c] = sc;
        mx = fmaxf(mx, sc);
      }
      mx = fmaxf(mx, __shfl_xor(mx, 16));
      float s = 0.f;
      __bf16* prow = pW + i * NPAD + half * 176;
      for (int c = 0; c < 176; ++c) {
        float e = __expf(srow[c] - mx);
        prow[c] = (__bf16)e;
        s += e;
      }
      s += __shfl_xor(s, 16);
      if (half == 0) dW[i] = s;
    }
    // out = P @ V  (two 16-wide d tiles, K = 352; single-base DS addressing)
    for (int dt = 0; dt < 2; ++dt) {
      v8f acc = {0.f, 0.f, 0.f, 0.f, 0.f, 0.f, 0.f, 0.f};
      const __bf16* vfrag0 = Vtsm + (dt * 16 + ln) * NPAD + half * 16;
      for (int kb = 0; kb < NPAD; kb += 32) {
        v8bf alo = *(const v8bf*)(pfrag0 + kb);
        v8bf ahi = *(const v8bf*)(pfrag0 + kb + 8);
        v8bf vlo = *(const v8bf*)(vfrag0 + kb);
        v8bf vhi = *(const v8bf*)(vfrag0 + kb + 8);
        acc = wmma_bf16(cat16(alo, ahi), cat16(vlo, vhi), acc);
      }
      int col = head * HDIM + dt * 16 + ln;
      for (int r = 0; r < 8; ++r) {
        int i = r + 8 * half, qn = q0 + i;
        if (qn < NTOK) {
          float dnm = dW[i];
          attnout[((long)win * NTOK + qn) * CCH + col] = (__bf16)(acc[r] / dnm);
        }
      }
    }
  }
}

// ---------------- launch ----------------
extern "C" void kernel_launch(void* const* d_in, const int* in_sizes, int n_in,
                              void* d_out, int out_size, void* d_ws, size_t ws_size,
                              hipStream_t stream) {
  const float* x      = (const float*)d_in[0];
  const float* mask   = (const float*)d_in[1];
  const float* skip   = (const float*)d_in[2];
  const float* x_up   = (const float*)d_in[3];
  const float* n1g    = (const float*)d_in[4];
  const float* n1b    = (const float*)d_in[5];
  const float* kv_w   = (const float*)d_in[6];
  const float* kv_b   = (const float*)d_in[7];
  const float* relb   = (const float*)d_in[8];
  const float* proj_w = (const float*)d_in[9];
  const float* proj_b = (const float*)d_in[10];
  const float* n2g    = (const float*)d_in[11];
  const float* n2b    = (const float*)d_in[12];
  const float* fc1_w  = (const float*)d_in[13];
  const float* fc1_b  = (const float*)d_in[14];
  const float* fc2_w  = (const float*)d_in[15];
  const float* fc2_b  = (const float*)d_in[16];
  float* out = (float*)d_out;

  char* ws = (char*)d_ws;
  __bf16* XU   = (__bf16*)(ws);                        // bf16 LTOK*192
  __bf16* SK   = (__bf16*)(ws + 16859136L);            // bf16 LTOK*192
  __bf16* KV   = (__bf16*)(ws + 33718272L);            // bf16 LTOK*384
  __bf16* AO   = (__bf16*)(ws + 67436544L);            // bf16 LTOK*192
  float*  X1   = (float*)(ws + 84295680L);             // f32  LTOK*192
  __bf16* H1   = (__bf16*)(ws + 118013952L);           // bf16 LTOK*192
  // region [134873088, 202309632) is time-shared:
  //   step 4 (attention): padded mask (128*343*352 f32 = 61.8 MB)
  //   steps 7-8 (MLP):    hidden activations (LTOK*768 bf16 = 67.4 MB)
  float*  MASKP = (float*)(ws + 134873088L);
  __bf16* HID  = (__bf16*)(ws + 134873088L);
  __bf16* WKV  = (__bf16*)(ws + 202309632L);           // bf16 384*192
  __bf16* WPRJ = (__bf16*)(ws + 202457088L);           // bf16 192*192
  __bf16* WF1  = (__bf16*)(ws + 202530816L);           // bf16 768*192
  __bf16* WF2  = (__bf16*)(ws + 202825728L);           // bf16 192*768
  float*  BIAS = (float*)(ws + 203120640L);            // f32  6*343*352
  (void)in_sizes; (void)n_in; (void)out_size; (void)ws_size;

  // 1. weight transpose -> bf16 ; rel-bias pre-gather ; mask padding
  prep_weights<<<1584, 256, 0, stream>>>(kv_w, proj_w, fc1_w, fc2_w, WKV, WPRJ, WF1, WF2);
  prep_bias<<<2831, 256, 0, stream>>>(relb, BIAS);
  prep_mask<<<60368, 256, 0, stream>>>(mask, MASKP);
  // 2. LN + shift + partition
  ln_shift_part<<<10976, 128, 0, stream>>>(skip, x_up, n1g, n1b, SK, XU);
  // 3. KV = sk_w @ kv_w + kv_b   (M=43904, N=384, K=192)
  gemm_bf16<0><<<8232, 256, 0, stream>>>(SK, WKV, kv_b, LTOK, 384, CCH, KV, nullptr, nullptr);
  // 4. attention per (window, head)
  attn_kernel<<<768, 128, 180480, stream>>>(XU, KV, BIAS, MASKP, AO);
  // 5. proj + reverse + roll + shortcut -> X1  (M=43904, N=192, K=192)
  gemm_bf16<3><<<4116, 256, 0, stream>>>(AO, WPRJ, proj_b, LTOK, CCH, CCH, nullptr, X1, x);
  // 6. LN2
  ln_plain<<<10976, 128, 0, stream>>>(X1, n2g, n2b, H1);
  // 7. fc1 + gelu  (M=43904, N=768, K=192)
  gemm_bf16<1><<<16464, 256, 0, stream>>>(H1, WF1, fc1_b, LTOK, HIDN, CCH, HID, nullptr, nullptr);
  // 8. fc2 + residual -> out  (M=43904, N=192, K=768)
  gemm_bf16<2><<<4116, 256, 0, stream>>>(HID, WF2, fc2_b, LTOK, CCH, HIDN, nullptr, out, X1);
}
